// InferModelBase_26345329394210
// MI455X (gfx1250) — matrix operand order repaired and checked by reading, not compiled, not run on hardware
//
#include <hip/hip_runtime.h>
#include <stdint.h>

#define E_EXPERTS 4
#define D0 384
#define D1 160
#define D2 128
#define D3 96
#define MBLK 64
#define THREADS 256
#define WAVES 8

typedef __attribute__((ext_vector_type(16))) __bf16 v16bf;
typedef __attribute__((ext_vector_type(8)))  float  v8f;

struct U128x2 { uint4 lo, hi; };

__device__ __forceinline__ unsigned short f2bf(float f) {
  unsigned int u = __builtin_bit_cast(unsigned int, f);
  u += 0x7FFFu + ((u >> 16) & 1u);            // round-to-nearest-even
  return (unsigned short)(u >> 16);
}
__device__ __forceinline__ float bf2f(unsigned short h) {
  unsigned int u = ((unsigned int)h) << 16;
  return __builtin_bit_cast(float, u);
}
__device__ __forceinline__ float celu01(float x) {
  // celu(x, alpha=0.1) = x>0 ? x : 0.1*(exp(x/0.1)-1)
  return x > 0.f ? x : 0.1f * (__expf(x * 10.f) - 1.f);
}

// ---------------- weight transpose + bf16 convert: Wt[e][dout][din] ----------
__global__ void k_wtrans(const float* __restrict__ W, unsigned short* __restrict__ Wt,
                         int din, int dout) {
  int total = E_EXPERTS * din * dout;
  for (int idx = blockIdx.x * blockDim.x + threadIdx.x; idx < total;
       idx += gridDim.x * blockDim.x) {
    int i   = idx % din;
    int rem = idx / din;
    int o   = rem % dout;
    int e   = rem / dout;
    Wt[idx] = f2bf(W[((size_t)e * din + i) * dout + o]);
  }
}

// ---------------- counting sort by species (stable, deterministic) ----------
__global__ void k_count(const int* __restrict__ sp, int* __restrict__ bc, int N) {
  __shared__ int c[E_EXPERTS];
  int t = threadIdx.x;
  if (t < E_EXPERTS) c[t] = 0;
  __syncthreads();
  int i = blockIdx.x * THREADS + t;
  if (i < N) atomicAdd(&c[sp[i]], 1);
  __syncthreads();
  if (t < E_EXPERTS) bc[blockIdx.x * E_EXPERTS + t] = c[t];
}

// one block; turns per-block counts into per-(block,species) start offsets,
// writes meta: [0..3]=cnt[e], [4..8]=speciesOff, [9..13]=tileBase (13=totalTiles)
__global__ void k_scan(int* __restrict__ bc, int* __restrict__ meta, int nblk) {
  extern __shared__ int s[];
  int* cnts = s;              // nblk*4
  int* aux  = s + nblk * 4;   // [0..3]=cnt, [4..7]=speciesOff
  int t = threadIdx.x;
  for (int i = t; i < nblk * 4; i += blockDim.x) cnts[i] = bc[i];
  __syncthreads();
  if (t < E_EXPERTS) {
    int run = 0;
    for (int b = 0; b < nblk; ++b) {
      int tmp = cnts[b * 4 + t];
      cnts[b * 4 + t] = run;
      run += tmp;
    }
    aux[t] = run;
  }
  __syncthreads();
  if (t == 0) {
    int off = 0, tb = 0;
    for (int e = 0; e < E_EXPERTS; ++e) {
      meta[e]     = aux[e];
      meta[4 + e] = off;
      meta[9 + e] = tb;
      aux[4 + e]  = off;
      off += aux[e];
      tb  += (aux[e] + MBLK - 1) / MBLK;
    }
    meta[8]  = off;
    meta[13] = tb;
  }
  __syncthreads();
  for (int i = t; i < nblk * 4; i += blockDim.x)
    bc[i] = cnts[i] + aux[4 + (i & 3)];
}

__global__ void k_scatter(const int* __restrict__ sp, const int* __restrict__ bc,
                          int* __restrict__ order, int N) {
  __shared__ int ssp[THREADS];
  int t = threadIdx.x;
  int base = blockIdx.x * THREADS;
  int i = base + t;
  ssp[t] = (i < N) ? sp[i] : -1;
  __syncthreads();
  if (t == 0) {
    int cur0 = bc[blockIdx.x * 4 + 0];
    int cur1 = bc[blockIdx.x * 4 + 1];
    int cur2 = bc[blockIdx.x * 4 + 2];
    int cur3 = bc[blockIdx.x * 4 + 3];
    int m = N - base; if (m > THREADS) m = THREADS;
    for (int j = 0; j < m; ++j) {       // in-order => stable
      int e = ssp[j];
      int pos;
      if      (e == 0) pos = cur0++;
      else if (e == 1) pos = cur1++;
      else if (e == 2) pos = cur2++;
      else             pos = cur3++;
      order[pos] = base + j;
    }
  }
}

// ---------------- WMMA MLP layer: dst[64][DOUT] = act(src[64][DIN] x Wt + b) --
// Register-blocked 2x2: each wave job computes a 32x32 output block with
// 4 accumulators, so each A (LDS) and B (global) fragment feeds 2 WMMAs.
// A (16x32 bf16) per ISA layout: lanes 0-15 rows M, K-chunks [k0+8h, k0+16+8h)
// B (32x16 bf16): lane column n=lane&15, K-chunks [k0+8h, k0+16+8h) matching A
// C/D (16x16 f32): vgpr i -> row i + 8*(lane>>4), col lane&15
template <int DIN, int DOUT, bool ACT>
__device__ __forceinline__ void mlp_layer(const unsigned short* __restrict__ src,
                                          unsigned short* __restrict__ dst,
                                          const unsigned short* __restrict__ Wt,
                                          const float* __restrict__ bias,
                                          int e, int wave, int lane) {
  const int NT2  = DOUT / 32;      // N-tile pairs
  const int JOBS = 2 * NT2;        // (mt pair) x (nt pair)
  const int half = lane >> 4;
  const int r    = lane & 15;
  for (int job = wave; job < JOBS; job += WAVES) {
    int mp = job & 1;              // m-pair: rows [mp*32, mp*32+32)
    int np = job >> 1;             // n-pair: cols [np*32, np*32+32)
    v8f acc00 = {}, acc01 = {}, acc10 = {}, acc11 = {};
    const unsigned short* arow0 = src + (mp * 32 + r) * DIN + half * 8;
    const unsigned short* arow1 = arow0 + 16 * DIN;
    const unsigned short* brow0 = Wt + (size_t)(e * DOUT + np * 32 + r) * DIN + half * 8;
    const unsigned short* brow1 = brow0 + (size_t)16 * DIN;
#pragma unroll 2
    for (int k0 = 0; k0 < DIN; k0 += 32) {
      U128x2 ua0, ua1, ub0, ub1;
      ua0.lo = *(const uint4*)(arow0 + k0);
      ua0.hi = *(const uint4*)(arow0 + k0 + 16);
      ua1.lo = *(const uint4*)(arow1 + k0);
      ua1.hi = *(const uint4*)(arow1 + k0 + 16);
      ub0.lo = *(const uint4*)(brow0 + k0);
      ub0.hi = *(const uint4*)(brow0 + k0 + 16);
      ub1.lo = *(const uint4*)(brow1 + k0);
      ub1.hi = *(const uint4*)(brow1 + k0 + 16);
      v16bf A0 = __builtin_bit_cast(v16bf, ua0);
      v16bf A1 = __builtin_bit_cast(v16bf, ua1);
      v16bf B0 = __builtin_bit_cast(v16bf, ub0);
      v16bf B1 = __builtin_bit_cast(v16bf, ub1);
      acc00 = __builtin_amdgcn_wmma_f32_16x16x32_bf16(false, A0, false, B0,
                                                      (short)0, acc00, false, false);
      acc01 = __builtin_amdgcn_wmma_f32_16x16x32_bf16(false, A0, false, B1,
                                                      (short)0, acc01, false, false);
      acc10 = __builtin_amdgcn_wmma_f32_16x16x32_bf16(false, A1, false, B0,
                                                      (short)0, acc10, false, false);
      acc11 = __builtin_amdgcn_wmma_f32_16x16x32_bf16(false, A1, false, B1,
                                                      (short)0, acc11, false, false);
    }
    float bv0 = bias[e * DOUT + np * 32 + r];
    float bv1 = bias[e * DOUT + np * 32 + 16 + r];
    unsigned short* d00 = dst + (mp * 32 + 8 * half) * DOUT + np * 32 + r;
#pragma unroll
    for (int i = 0; i < 8; ++i) {
      float x00 = acc00[i] + bv0;
      float x01 = acc01[i] + bv1;
      float x10 = acc10[i] + bv0;
      float x11 = acc11[i] + bv1;
      if (ACT) { x00 = celu01(x00); x01 = celu01(x01); x10 = celu01(x10); x11 = celu01(x11); }
      d00[i * DOUT]               = f2bf(x00);
      d00[i * DOUT + 16]          = f2bf(x01);
      d00[(16 + i) * DOUT]        = f2bf(x10);
      d00[(16 + i) * DOUT + 16]   = f2bf(x11);
    }
  }
}

// ---------------- main grouped-MoE kernel: 64 atoms (one species) per block --
__global__ void __launch_bounds__(THREADS)
k_moe(const float* __restrict__ aev, const int* __restrict__ order,
      const int* __restrict__ meta,
      const unsigned short* __restrict__ Wt1, const unsigned short* __restrict__ Wt2,
      const unsigned short* __restrict__ Wt3, const unsigned short* __restrict__ Wt4,
      const float* __restrict__ b1, const float* __restrict__ b2,
      const float* __restrict__ b3, const float* __restrict__ b4,
      float* __restrict__ partials) {
  __shared__ __align__(16) unsigned short sA[MBLK * D0];   // 49152 B
  __shared__ __align__(16) unsigned short sB[MBLK * D1];   // 20480 B
  __shared__ int   sIdx[MBLK];
  __shared__ int   sMeta[16];
  __shared__ float sRed[THREADS];
  int t = threadIdx.x;
  if (t < 14) sMeta[t] = meta[t];
  __syncthreads();
  int total = sMeta[13];
  int blk = blockIdx.x;
  if (blk >= total) { if (t == 0) partials[blk] = 0.f; return; }  // uniform exit
  int e = 0;
  while (e < E_EXPERTS - 1 && blk >= sMeta[9 + e + 1]) ++e;
  int tile  = blk - sMeta[9 + e];
  int soff  = sMeta[4 + e];
  int cnt   = sMeta[e];
  int valid = cnt - tile * MBLK; if (valid > MBLK) valid = MBLK;
  if (t < MBLK) sIdx[t] = (t < valid) ? order[soff + tile * MBLK + t] : 0;
  __syncthreads();

  // gather AEV rows, fp32 -> bf16 into LDS (float4 granularity)
  for (int c = t; c < MBLK * (D0 / 4); c += THREADS) {
    int m  = c / (D0 / 4);
    int k4 = c % (D0 / 4);
    float4 v = make_float4(0.f, 0.f, 0.f, 0.f);
    if (m < valid) v = *(const float4*)(aev + (size_t)sIdx[m] * D0 + k4 * 4);
    unsigned int p0 = (unsigned int)f2bf(v.x) | ((unsigned int)f2bf(v.y) << 16);
    unsigned int p1 = (unsigned int)f2bf(v.z) | ((unsigned int)f2bf(v.w) << 16);
    *(uint2*)(sA + m * D0 + k4 * 4) = make_uint2(p0, p1);
  }
  __syncthreads();

  int wave = t >> 5, lane = t & 31;
  mlp_layer<D0, D1, true>(sA, sB, Wt1, b1, e, wave, lane);   // x  -> h1
  __syncthreads();
  mlp_layer<D1, D2, true>(sB, sA, Wt2, b2, e, wave, lane);   // h1 -> h2
  __syncthreads();
  mlp_layer<D2, D3, true>(sA, sB, Wt3, b3, e, wave, lane);   // h2 -> h3
  __syncthreads();

  // layer 4: per-atom dot (dout = 1), mask padded rows, block reduce
  int a = t >> 2, p = t & 3;
  float s = 0.f;
  const unsigned short* w4 = Wt4 + e * D3;
  for (int j = p * 24; j < p * 24 + 24; ++j)
    s += bf2f(sB[a * D3 + j]) * bf2f(w4[j]);
  sRed[t] = s;
  __syncthreads();
  if (t < MBLK) {
    float v = sRed[4 * t] + sRed[4 * t + 1] + sRed[4 * t + 2] + sRed[4 * t + 3] + b4[e];
    sRed[t] = (t < valid) ? v : 0.f;
  }
  __syncthreads();
  if (t < 32) sRed[t] += sRed[t + 32];
  __syncthreads();
  if (t == 0) {
    float sum = 0.f;
    for (int i = 0; i < 32; ++i) sum += sRed[i];
    partials[blk] = sum;
  }
}

// ---------------- fixed-order final reduction -------------------------------
__global__ void k_reduce(const float* __restrict__ partials, int n,
                         float* __restrict__ out) {
  __shared__ float s[THREADS];
  int t = threadIdx.x;
  float v = 0.f;
  for (int i = t; i < n; i += THREADS) v += partials[i];
  s[t] = v;
  __syncthreads();
  for (int w = THREADS / 2; w > 0; w >>= 1) {
    if (t < w) s[t] += s[t + w];
    __syncthreads();
  }
  if (t == 0) out[0] = s[0];
}

extern "C" void kernel_launch(void* const* d_in, const int* in_sizes, int n_in,
                              void* d_out, int out_size, void* d_ws, size_t ws_size,
                              hipStream_t stream) {
  const float* aev = (const float*)d_in[0];
  const int*   sp  = (const int*)d_in[1];
  const float* W1  = (const float*)d_in[2];
  const float* b1  = (const float*)d_in[3];
  const float* W2  = (const float*)d_in[4];
  const float* b2  = (const float*)d_in[5];
  const float* W3  = (const float*)d_in[6];
  const float* b3  = (const float*)d_in[7];
  const float* W4  = (const float*)d_in[8];
  const float* b4  = (const float*)d_in[9];

  int N     = in_sizes[1];
  int nblk  = (N + THREADS - 1) / THREADS;
  int ntile = (N + MBLK - 1) / MBLK + E_EXPERTS;   // padded grid for main kernel

  char* ws = (char*)d_ws;
  size_t off = 0;
  auto alloc = [&](size_t bytes) -> void* {
    void* p = ws + off;
    off += (bytes + 255) & ~(size_t)255;
    return p;
  };
  int*   order    = (int*)alloc((size_t)N * 4);
  int*   bc       = (int*)alloc((size_t)nblk * 4 * 4);
  int*   meta     = (int*)alloc(64);
  float* partials = (float*)alloc((size_t)ntile * 4);
  unsigned short* Wt1 = (unsigned short*)alloc((size_t)E_EXPERTS * D1 * D0 * 2);
  unsigned short* Wt2 = (unsigned short*)alloc((size_t)E_EXPERTS * D2 * D1 * 2);
  unsigned short* Wt3 = (unsigned short*)alloc((size_t)E_EXPERTS * D3 * D2 * 2);
  unsigned short* Wt4 = (unsigned short*)alloc((size_t)E_EXPERTS * 1  * D3 * 2);

  k_wtrans<<<256, THREADS, 0, stream>>>(W1, Wt1, D0, D1);
  k_wtrans<<<256, THREADS, 0, stream>>>(W2, Wt2, D1, D2);
  k_wtrans<<<128, THREADS, 0, stream>>>(W3, Wt3, D2, D3);
  k_wtrans<<<4,   THREADS, 0, stream>>>(W4, Wt4, D3, 1);

  k_count<<<nblk, THREADS, 0, stream>>>(sp, bc, N);
  k_scan<<<1, THREADS, (size_t)(nblk * 4 + 16) * sizeof(int), stream>>>(bc, meta, nblk);
  k_scatter<<<nblk, THREADS, 0, stream>>>(sp, bc, order, N);

  k_moe<<<ntile, THREADS, 0, stream>>>(aev, order, meta, Wt1, Wt2, Wt3, Wt4,
                                       b1, b2, b3, b4, partials);
  k_reduce<<<1, THREADS, 0, stream>>>(partials, ntile, (float*)d_out);
}
